// MultiHeadSelfAttention_28604482191479
// MI455X (gfx1250) — matrix operand order repaired and checked by reading, not compiled, not run on hardware
//
#include <hip/hip_runtime.h>
#include <hip/hip_bf16.h>
#include <cmath>

typedef float    v2f  __attribute__((ext_vector_type(2)));
typedef float    v8f  __attribute__((ext_vector_type(8)));
typedef _Float16 v8h  __attribute__((ext_vector_type(8)));
typedef _Float16 v16h __attribute__((ext_vector_type(16)));

constexpr int B  = 32;
constexpr int T  = 2048;
constexpr int H  = 1024;
constexpr int NH = 8;
constexpr int HD = 128;   // H / NH
constexpr int R  = B * T; // 65536 rows of X

static __device__ __forceinline__ v8f wmma4(v2f a, v2f b, v8f c) {
    return __builtin_amdgcn_wmma_f32_16x16x4_f32(
        false, a, false, b, (short)0, c, false, false);
}
static __device__ __forceinline__ v8f wmma32(v16h a, v16h b, v8f c) {
    return __builtin_amdgcn_wmma_f32_16x16x32_f16(
        false, a, false, b, (short)0, c, false, false);
}

union V16 { v16h v; v8h h[2]; };

// ---------------------------------------------------------------------------
// Kernel W: transpose + convert W_attn (NH,H,HD) f32 -> Wt (NH,HD,H) f16.
// 4 MB once per launch; makes the WMMA B-operand (K contiguous) two b128 loads.
// ---------------------------------------------------------------------------
__global__ __launch_bounds__(256) void wconv_kernel(
    const float* __restrict__ Wa, _Float16* __restrict__ Wt)
{
    __shared__ float tile[32][33];
    const int n  = blockIdx.x;
    const int h0 = blockIdx.y * 32;
    const int d0 = blockIdx.z * 32;
    const int tx = threadIdx.x & 31;
    const int ty = threadIdx.x >> 5;
#pragma unroll
    for (int i = 0; i < 4; ++i)
        tile[ty + 8 * i][tx] = Wa[((size_t)n * H + h0 + ty + 8 * i) * HD + d0 + tx];
    __syncthreads();
#pragma unroll
    for (int i = 0; i < 4; ++i)
        Wt[((size_t)n * HD + d0 + ty + 8 * i) * H + h0 + tx] =
            (_Float16)tile[tx][ty + 8 * i];
}

// ---------------------------------------------------------------------------
// Kernel A: fused scores[b,n,t] = sum_d tanh((x_row . W[:,d]) + b[d]) * v[d]
// Block = 256 threads (8 waves) -> 64 rows x 2 heads (256 cols).
// Wave w: head hl=w>>2, columns (w&3)*32 .. +32 (2 N-tiles, 8 accumulators).
// x slab is converted f32->f16 into LDS each K=32 chunk (no extra HBM pass).
// ---------------------------------------------------------------------------
__global__ __launch_bounds__(256) void score_kernel(
    const float*    __restrict__ x,      // (B,T,H) f32
    const _Float16* __restrict__ Wt,     // (NH,HD,H) f16
    const float*    __restrict__ ba,     // (NH,HD)
    const float*    __restrict__ vs,     // (NH,HD)
    float*          __restrict__ scores) // (B,NH,T)
{
    constexpr int AST = 40;                       // LDS row stride in halves
    __shared__ __align__(16) _Float16 As[64 * AST];
    __shared__ float sSc[2][64];

    const size_t rowBase = (size_t)blockIdx.x * 64;
    const int tid = threadIdx.x;
    const int w   = tid >> 5;
    const int l   = tid & 31;
    const int lm  = l & 15;
    const int lh  = l >> 4;
    const int hl  = w >> 2;                       // head within block (0/1)
    const int n   = blockIdx.y * 2 + hl;          // global head
    const int cb  = (w & 3) * 32;                 // col base within head

    const _Float16* Wp0 = Wt + ((size_t)n * HD + cb + lm) * H;        // N-tile 0
    const _Float16* Wp1 = Wp0 + (size_t)16 * H;                       // N-tile 1

    v8f acc[8] = {};                              // [rt][ct] -> rt*2+ct

    if (tid < 128) sSc[tid >> 6][tid & 63] = 0.0f;

    // cooperative x-slab mapping: thread -> (row, 8-float column group)
    const int ar  = tid >> 2;                     // 0..63
    const int akk = (tid & 3) << 3;               // 0,8,16,24

    for (int k0 = 0; k0 < H; k0 += 32) {
        __syncthreads();
        {
            const float* src = x + (rowBase + ar) * H + k0 + akk;
            float4 f0 = *(const float4*)src;
            float4 f1 = *(const float4*)(src + 4);
            v8h hv;
            hv[0] = (_Float16)f0.x; hv[1] = (_Float16)f0.y;
            hv[2] = (_Float16)f0.z; hv[3] = (_Float16)f0.w;
            hv[4] = (_Float16)f1.x; hv[5] = (_Float16)f1.y;
            hv[6] = (_Float16)f1.z; hv[7] = (_Float16)f1.w;
            *(v8h*)&As[ar * AST + akk] = hv;
            if (k0 + 32 < H) __builtin_prefetch(src + 32, 0, 0);
        }
        __syncthreads();

        // B operands: 32x16 f16, lane = column lm, K = [lh*16, lh*16+16)
        V16 b0, b1;
        b0.v = *(const v16h*)(Wp0 + k0 + lh * 16);
        b1.v = *(const v16h*)(Wp1 + k0 + lh * 16);

#pragma unroll
        for (int rt = 0; rt < 4; ++rt) {
            // A operand: 16x32 f16, lane row = lm, K = [lh*16, lh*16+16)
            const _Float16* arow = &As[(rt * 16 + lm) * AST + lh * 16];
            V16 a;
            a.h[0] = *(const v8h*)(arow);
            a.h[1] = *(const v8h*)(arow + 8);
            acc[rt * 2 + 0] = wmma32(a.v, b0.v, acc[rt * 2 + 0]);
            acc[rt * 2 + 1] = wmma32(a.v, b1.v, acc[rt * 2 + 1]);
        }
    }

    // Epilogue: tanh(acc + bias) * v, reduce this wave's 32 cols per row.
    const float bb0 = ba[n * HD + cb + lm],      vv0 = vs[n * HD + cb + lm];
    const float bb1 = ba[n * HD + cb + 16 + lm], vv1 = vs[n * HD + cb + 16 + lm];
#pragma unroll
    for (int rt = 0; rt < 4; ++rt) {
#pragma unroll
        for (int rr = 0; rr < 8; ++rr) {
            float e = tanhf(acc[rt * 2 + 0][rr] + bb0) * vv0 +
                      tanhf(acc[rt * 2 + 1][rr] + bb1) * vv1;
            e += __shfl_xor(e, 8, 16);
            e += __shfl_xor(e, 4, 16);
            e += __shfl_xor(e, 2, 16);
            e += __shfl_xor(e, 1, 16);
            if (lm == 0) atomicAdd(&sSc[hl][rt * 16 + rr + lh * 8], e);
        }
    }
    __syncthreads();
    if (tid < 128) {
        const int hl2 = tid >> 6, loc = tid & 63;
        const size_t row = rowBase + loc;
        const int bidx = (int)(row / T);
        const int tt   = (int)(row % T);
        scores[((size_t)bidx * NH + blockIdx.y * 2 + hl2) * T + tt] = sSc[hl2][loc];
    }
}

// ---------------------------------------------------------------------------
// Kernel B: softmax over T per (b,n) row; writes attn (second output).
// ---------------------------------------------------------------------------
__global__ __launch_bounds__(256) void softmax_kernel(
    const float* __restrict__ scores, float* __restrict__ attn)
{
    __shared__ float red[256];
    const int row = blockIdx.x;
    const int tid = threadIdx.x;
    const float* s = scores + (size_t)row * T;
    float* a = attn + (size_t)row * T;

    float m = -3.402823466e38f;
    for (int i = tid; i < T; i += 256) m = fmaxf(m, s[i]);
    red[tid] = m; __syncthreads();
    for (int off = 128; off > 0; off >>= 1) {
        if (tid < off) red[tid] = fmaxf(red[tid], red[tid + off]);
        __syncthreads();
    }
    m = red[0];
    __syncthreads();

    float sum = 0.0f;
    for (int i = tid; i < T; i += 256) {
        float e = expf(s[i] - m);
        a[i] = e;
        sum += e;
    }
    red[tid] = sum; __syncthreads();
    for (int off = 128; off > 0; off >>= 1) {
        if (tid < off) red[tid] += red[tid + off];
        __syncthreads();
    }
    const float inv = 1.0f / red[0];
    for (int i = tid; i < T; i += 256) a[i] *= inv;
}

// ---------------------------------------------------------------------------
// Kernel C: ctx[b,n,d] = sum_t attn[b,n,t] * x[b,t,d]   (d < HD)
// ---------------------------------------------------------------------------
__global__ __launch_bounds__(128) void ctx_kernel(
    const float* __restrict__ x, const float* __restrict__ attn,
    float* __restrict__ ctx)
{
    __shared__ float at[NH][128];
    const int b = blockIdx.x;
    const int d = threadIdx.x;
    float acc[NH] = {};

    for (int t0 = 0; t0 < T; t0 += 128) {
        __syncthreads();
#pragma unroll
        for (int i = 0; i < NH; ++i)
            at[i][d] = attn[((size_t)b * NH + i) * T + t0 + d];
        __syncthreads();
        for (int tt = 0; tt < 128; ++tt) {
            float xv = x[((size_t)b * T + t0 + tt) * H + d];
#pragma unroll
            for (int i = 0; i < NH; ++i) acc[i] += at[i][tt] * xv;
        }
    }
#pragma unroll
    for (int i = 0; i < NH; ++i)
        ctx[(size_t)b * H + i * HD + d] = acc[i];
}

// ---------------------------------------------------------------------------
// Kernel D: out(32x1024) = ctx @ W_out + b_out via exact f32 WMMA (tiny).
// ---------------------------------------------------------------------------
__global__ __launch_bounds__(32) void outproj_kernel(
    const float* __restrict__ ctx, const float* __restrict__ Wo,
    const float* __restrict__ bo, float* __restrict__ out)
{
    const int ct = blockIdx.x;
    const int l  = threadIdx.x;
    const int lm = l & 15;
    const int lh = l >> 4;
    const int col = ct * 16 + lm;

    v8f acc[2] = {};
    for (int k0 = 0; k0 < H; k0 += 4) {
        const int k = k0 + lh * 2;
        v2f bv;
        bv.x = Wo[(size_t)k * H + col];
        bv.y = Wo[(size_t)(k + 1) * H + col];
        v2f a0 = *(const v2f*)&ctx[(size_t)lm * H + k];
        v2f a1 = *(const v2f*)&ctx[(size_t)(lm + 16) * H + k];
        acc[0] = wmma4(a0, bv, acc[0]);
        acc[1] = wmma4(a1, bv, acc[1]);
    }
    const float bb = bo[col];
#pragma unroll
    for (int rt = 0; rt < 2; ++rt)
#pragma unroll
        for (int rr = 0; rr < 8; ++rr) {
            const int m = rt * 16 + rr + lh * 8;
            out[(size_t)m * H + col] = acc[rt][rr] + bb;
        }
}

// ---------------------------------------------------------------------------
extern "C" void kernel_launch(void* const* d_in, const int* in_sizes, int n_in,
                              void* d_out, int out_size, void* d_ws, size_t ws_size,
                              hipStream_t stream) {
    const float* x  = (const float*)d_in[0];   // (B,T,H)
    const float* Wa = (const float*)d_in[1];   // (NH,H,HD)
    const float* ba = (const float*)d_in[2];   // (NH,HD)
    const float* vs = (const float*)d_in[3];   // (NH,HD)
    const float* Wo = (const float*)d_in[4];   // (H,H)
    const float* bo = (const float*)d_in[5];   // (H,)

    float* out_ctx  = (float*)d_out;                     // B*H
    float* out_attn = out_ctx + (size_t)B * H;           // B*NH*T

    float*     ws     = (float*)d_ws;
    float*     scores = ws;                              // B*NH*T floats
    float*     ctx    = ws + (size_t)B * NH * T;         // B*H floats
    _Float16*  Wt     = (_Float16*)(ctx + (size_t)B * H); // NH*HD*H halves (2 MB)

    wconv_kernel<<<dim3(NH, H / 32, HD / 32), 256, 0, stream>>>(Wa, Wt);

    dim3 gA(R / 64, NH / 2);
    score_kernel<<<gA, 256, 0, stream>>>(x, Wt, ba, vs, scores);
    softmax_kernel<<<B * NH, 256, 0, stream>>>(scores, out_attn);
    ctx_kernel<<<B, 128, 0, stream>>>(x, out_attn, ctx);
    outproj_kernel<<<H / 16, 32, 0, stream>>>(ctx, Wo, bo, out_ctx);
}
